// MLPSepConv_84207128805580
// MI455X (gfx1250) — 8-bit matrix operand order repaired and checked by reading, compiled, not run on hardware
//
#include <hip/hip_runtime.h>
#include <cstdint>

typedef __attribute__((ext_vector_type(8))) int v8i;

#define QEPS 1e-5f

#if defined(__HIP_DEVICE_COMPILE__) && \
    __has_builtin(__builtin_amdgcn_global_load_async_to_lds_b128) && \
    __has_builtin(__builtin_amdgcn_s_wait_asynccnt)
#define USE_ASYNC_LDS 1
#else
#define USE_ASYNC_LDS 0
#endif

#if USE_ASYNC_LDS
typedef int async_v4i __attribute__((vector_size(16)));
typedef __attribute__((address_space(1))) async_v4i as1_v4i;
typedef __attribute__((address_space(3))) async_v4i as3_v4i;
#endif

__global__ __launch_bounds__(256) void k_abs_partial(const float* __restrict__ w, int n,
                                                     float* __restrict__ partials) {
  __shared__ float sm[256];
  float s = 0.f;
  for (int i = blockIdx.x * 256 + threadIdx.x; i < n; i += gridDim.x * 256)
    s += fabsf(w[i]);
  sm[threadIdx.x] = s;
  __syncthreads();
  for (int st = 128; st > 0; st >>= 1) {
    if (threadIdx.x < st) sm[threadIdx.x] += sm[threadIdx.x + st];
    __syncthreads();
  }
  if (threadIdx.x == 0) partials[blockIdx.x] = sm[0];
}

__global__ __launch_bounds__(256) void k_finalize_scale(const float* __restrict__ partials,
                                                        int np, float inv_count,
                                                        float* __restrict__ wscale) {
  __shared__ float sm[256];
  float s = 0.f;
  for (int i = threadIdx.x; i < np; i += 256) s += partials[i];
  sm[threadIdx.x] = s;
  __syncthreads();
  for (int st = 128; st > 0; st >>= 1) {
    if (threadIdx.x < st) sm[threadIdx.x] += sm[threadIdx.x + st];
    __syncthreads();
  }
  if (threadIdx.x == 0) wscale[0] = fmaxf(sm[0] * inv_count, QEPS);
}

__global__ __launch_bounds__(256) void k_quant_weight(const float* __restrict__ w,
                                                      const float* __restrict__ wscale,
                                                      int8_t* __restrict__ wq, int n) {
  const float s = 1.0f / wscale[0];
  for (int i = blockIdx.x * 256 + threadIdx.x; i < n; i += gridDim.x * 256) {
    float q = rintf(w[i] * s);
    q = fminf(fmaxf(q, -1.f), 1.f);
    wq[i] = (int8_t)q;
  }
}

__global__ __launch_bounds__(256) void k_actquant_rows(const float* __restrict__ x,
                                                       int8_t* __restrict__ xq,
                                                       float* __restrict__ inv_s) {
  const int m = blockIdx.x;
  const int tid = threadIdx.x;
  const float* row = x + (size_t)m * 768;
  const float v0 = row[tid], v1 = row[tid + 256], v2 = row[tid + 512];

  __shared__ float ssum[256], smax[256], bc[1];
  ssum[tid] = v0 * v0 + v1 * v1 + v2 * v2;
  smax[tid] = fmaxf(fabsf(v0), fmaxf(fabsf(v1), fabsf(v2)));
  __syncthreads();
  for (int st = 128; st > 0; st >>= 1) {
    if (tid < st) {
      ssum[tid] += ssum[tid + st];
      smax[tid] = fmaxf(smax[tid], smax[tid + st]);
    }
    __syncthreads();
  }
  if (tid == 0) {
    const float rms = rsqrtf(ssum[0] * (1.0f / 768.0f) + 1e-6f);
    const float am = fmaxf(smax[0] * rms, QEPS);
    bc[0] = rms * (127.0f / am);
    inv_s[m] = am / 127.0f;
  }
  __syncthreads();
  const float sc = bc[0];
  int8_t* orow = xq + (size_t)m * 768;
  float q0 = fminf(fmaxf(rintf(v0 * sc), -128.f), 127.f);
  float q1 = fminf(fmaxf(rintf(v1 * sc), -128.f), 127.f);
  float q2 = fminf(fmaxf(rintf(v2 * sc), -128.f), 127.f);
  orow[tid]       = (int8_t)q0;
  orow[tid + 256] = (int8_t)q1;
  orow[tid + 512] = (int8_t)q2;
}

__device__ __forceinline__ v8i load_a_frag(const int8_t* __restrict__ ap) {
  const int4 t0 = *(const int4*)(ap);
  const int4 t1 = *(const int4*)(ap + 32);
  v8i a;
  a[0] = t0.x; a[1] = t0.y; a[2] = t0.z; a[3] = t0.w;
  a[4] = t1.x; a[5] = t1.y; a[6] = t1.z; a[7] = t1.w;
  return a;
}

__device__ __forceinline__ v8i load_b_frag(const int8_t* __restrict__ bp) {
  const int4 u0 = *(const int4*)(bp);
  const int4 u1 = *(const int4*)(bp + 32);
  v8i b;
  b[0] = u0.x; b[1] = u0.y; b[2] = u0.z; b[3] = u0.w;
  b[4] = u1.x; b[5] = u1.y; b[6] = u1.z; b[7] = u1.w;
  return b;
}

__global__ __launch_bounds__(256) void k_gemm_iu8(const int8_t* __restrict__ xq,
                                                  const int8_t* __restrict__ wq,
                                                  const float* __restrict__ inv_sx,
                                                  const float* __restrict__ wscale,
                                                  const float* __restrict__ bias,
                                                  float* __restrict__ out,
                                                  int K, int N, int nchw) {
  const int tid = threadIdx.x;
  const int lane = tid & 31, wave = tid >> 5;
  const int l16 = lane & 15, lh = lane >> 4;
  const int m_base = blockIdx.x * 32 + (wave & 1) * 16;
  const int n_base = blockIdx.y * 256 + (wave >> 1) * 64;

  v8i acc[4] = {};

  const int8_t* aptr = xq + (size_t)(m_base + l16) * K + (lh << 4);
  const int8_t* bptr = wq + (size_t)(n_base + l16) * K + (lh << 4);
  const size_t bstride = (size_t)16 * K;

  v8i a_cur  = load_a_frag(aptr);
  v8i b_cur0 = load_b_frag(bptr);
  v8i b_cur1 = load_b_frag(bptr + bstride);
  v8i b_cur2 = load_b_frag(bptr + 2 * bstride);
  v8i b_cur3 = load_b_frag(bptr + 3 * bstride);

  for (int k0 = 64; k0 < K; k0 += 64) {
    v8i a_nxt  = load_a_frag(aptr + k0);
    v8i b_nxt0 = load_b_frag(bptr + k0);
    v8i b_nxt1 = load_b_frag(bptr + bstride + k0);
    v8i b_nxt2 = load_b_frag(bptr + 2 * bstride + k0);
    v8i b_nxt3 = load_b_frag(bptr + 3 * bstride + k0);

    acc[0] = __builtin_amdgcn_wmma_i32_16x16x64_iu8(true, a_cur, true, b_cur0, acc[0], false, false);
    acc[1] = __builtin_amdgcn_wmma_i32_16x16x64_iu8(true, a_cur, true, b_cur1, acc[1], false, false);
    acc[2] = __builtin_amdgcn_wmma_i32_16x16x64_iu8(true, a_cur, true, b_cur2, acc[2], false, false);
    acc[3] = __builtin_amdgcn_wmma_i32_16x16x64_iu8(true, a_cur, true, b_cur3, acc[3], false, false);

    a_cur = a_nxt;
    b_cur0 = b_nxt0; b_cur1 = b_nxt1; b_cur2 = b_nxt2; b_cur3 = b_nxt3;
  }

  acc[0] = __builtin_amdgcn_wmma_i32_16x16x64_iu8(true, a_cur, true, b_cur0, acc[0], false, false);
  acc[1] = __builtin_amdgcn_wmma_i32_16x16x64_iu8(true, a_cur, true, b_cur1, acc[1], false, false);
  acc[2] = __builtin_amdgcn_wmma_i32_16x16x64_iu8(true, a_cur, true, b_cur2, acc[2], false, false);
  acc[3] = __builtin_amdgcn_wmma_i32_16x16x64_iu8(true, a_cur, true, b_cur3, acc[3], false, false);

  const float wsc = wscale[0];
  const int mrow = m_base + (lh << 3);
  float scl[8];
#pragma unroll
  for (int r = 0; r < 8; ++r) scl[r] = inv_sx[mrow + r] * wsc;

#pragma unroll
  for (int j = 0; j < 4; ++j) {
    const int nj = n_base + j * 16 + l16;
    const float bj = bias[nj];
    float f[8];
#pragma unroll
    for (int r = 0; r < 8; ++r) f[r] = (float)acc[j][r] * scl[r] + bj;
    if (nchw) {
      const int bidx = mrow >> 10, t = mrow & 1023;
      float* o = out + ((size_t)bidx * N + nj) * 1024 + t;
      *(float4*)(o)     = make_float4(f[0], f[1], f[2], f[3]);
      *(float4*)(o + 4) = make_float4(f[4], f[5], f[6], f[7]);
    } else {
#pragma unroll
      for (int r = 0; r < 8; ++r) out[(size_t)(mrow + r) * N + nj] = f[r];
    }
  }
}

__global__ __launch_bounds__(256) void k_dwconv_gelu(const float* __restrict__ cw,
                                                     const float* __restrict__ cb,
                                                     float* __restrict__ y, int C) {
  __shared__ float sm[1024];
  const int bc = blockIdx.x;
  const int c = bc % C;
  float* plane = y + (size_t)bc * 1024;
  const int tid = threadIdx.x;

#if USE_ASYNC_LDS
  {
    __builtin_amdgcn_global_load_async_to_lds_b128(
        (as1_v4i*)(plane + tid * 4), (as3_v4i*)(sm + tid * 4), 0, 0);
    __builtin_amdgcn_s_wait_asynccnt(0);
  }
#else
  for (int i = tid; i < 1024; i += 256) sm[i] = plane[i];
#endif
  __syncthreads();

  float wk[9];
#pragma unroll
  for (int q = 0; q < 9; ++q) wk[q] = cw[c * 9 + q];
  const float bb = cb[c];

  float res[4];
#pragma unroll
  for (int p = 0; p < 4; ++p) {
    const int idx = tid + p * 256;
    const int h = idx >> 5, w = idx & 31;
    float acc = bb;
#pragma unroll
    for (int kh = 0; kh < 3; ++kh)
#pragma unroll
      for (int kw = 0; kw < 3; ++kw) {
        const int hh = h + kh - 1, ww = w + kw - 1;
        if (hh >= 0 && hh < 32 && ww >= 0 && ww < 32)
          acc += sm[(hh << 5) + ww] * wk[kh * 3 + kw];
      }
    res[p] = 0.5f * acc * (1.0f + erff(acc * 0.70710678118654752f));
  }
#pragma unroll
  for (int p = 0; p < 4; ++p) plane[tid + p * 256] = res[p];
}

__global__ __launch_bounds__(256) void k_actquant_nchw(const float* __restrict__ y,
                                                       int8_t* __restrict__ xq,
                                                       float* __restrict__ inv_s) {
  __shared__ float tile[3072 * 8];
  __shared__ float rsum[256], rmax[256], sfac[8];
  const int tid = threadIdx.x;
  const int b = blockIdx.x >> 7;
  const int t0 = (blockIdx.x & 127) << 3;
  const float* src = y + (size_t)b * 3072 * 1024 + t0;

  const int tt = tid & 7, cidx = tid >> 3;
  for (int c = cidx; c < 3072; c += 32)
    tile[(c << 3) + tt] = src[(size_t)c * 1024 + tt];
  __syncthreads();

  float ssq = 0.f, amx = 0.f;
  for (int c = cidx; c < 3072; c += 32) {
    const float v = tile[(c << 3) + tt];
    ssq += v * v;
    amx = fmaxf(amx, fabsf(v));
  }
  rsum[tid] = ssq;
  rmax[tid] = amx;
  __syncthreads();
  for (int st = 128; st >= 8; st >>= 1) {
    if (tid < st) {
      rsum[tid] += rsum[tid + st];
      rmax[tid] = fmaxf(rmax[tid], rmax[tid + st]);
    }
    __syncthreads();
  }
  if (tid < 8) {
    const float rms = rsqrtf(rsum[tid] * (1.0f / 3072.0f) + 1e-6f);
    const float am = fmaxf(rmax[tid] * rms, QEPS);
    sfac[tid] = rms * (127.0f / am);
    inv_s[((size_t)b << 10) + t0 + tid] = am / 127.0f;
  }
  __syncthreads();

  const int wt = tid >> 5;
  const float sc = sfac[wt];
  int8_t* dst = xq + (size_t)(((size_t)b << 10) + t0 + wt) * 3072;
  for (int c = tid & 31; c < 3072; c += 32) {
    float q = rintf(tile[(c << 3) + wt] * sc);
    q = fminf(fmaxf(q, -128.f), 127.f);
    dst[c] = (int8_t)q;
  }
}

extern "C" void kernel_launch(void* const* d_in, const int* in_sizes, int n_in,
                              void* d_out, int out_size, void* d_ws, size_t ws_size,
                              hipStream_t stream) {
  (void)in_sizes; (void)n_in; (void)out_size; (void)ws_size;
  const float* x  = (const float*)d_in[0];
  const float* w1 = (const float*)d_in[1];
  const float* b1 = (const float*)d_in[2];
  const float* cw = (const float*)d_in[3];
  const float* cb = (const float*)d_in[4];
  const float* w2 = (const float*)d_in[5];
  const float* b2 = (const float*)d_in[6];
  float* out = (float*)d_out;

  const int D = 768, H = 3072, M = 32768;

  char* ws = (char*)d_ws;
  size_t off = 0;
  auto alloc = [&](size_t bytes) -> char* {
    char* p = ws + off;
    off += (bytes + 255) & ~(size_t)255;
    return p;
  };
  int8_t* XQ1  = (int8_t*)alloc((size_t)M * D);
  float*  IS1  = (float*)alloc((size_t)M * 4);
  int8_t* XQ2  = (int8_t*)alloc((size_t)M * H);
  float*  IS2  = (float*)alloc((size_t)M * 4);
  int8_t* WQ1  = (int8_t*)alloc((size_t)H * D);
  int8_t* WQ2  = (int8_t*)alloc((size_t)D * H);
  float*  WS1  = (float*)alloc(256);
  float*  WS2  = (float*)alloc(256);
  float*  PART = (float*)alloc(512 * 4);
  float*  Y    = (float*)alloc((size_t)M * H * 4);

  k_abs_partial<<<512, 256, 0, stream>>>(w1, H * D, PART);
  k_finalize_scale<<<1, 256, 0, stream>>>(PART, 512, 1.0f / (float)(H * D), WS1);
  k_quant_weight<<<512, 256, 0, stream>>>(w1, WS1, WQ1, H * D);
  k_abs_partial<<<512, 256, 0, stream>>>(w2, D * H, PART);
  k_finalize_scale<<<1, 256, 0, stream>>>(PART, 512, 1.0f / (float)(D * H), WS2);
  k_quant_weight<<<512, 256, 0, stream>>>(w2, WS2, WQ2, D * H);

  k_actquant_rows<<<M, 256, 0, stream>>>(x, XQ1, IS1);

  k_gemm_iu8<<<dim3(M / 32, H / 256), 256, 0, stream>>>(XQ1, WQ1, IS1, WS1, b1, Y,
                                                        D, H, 1);
  k_dwconv_gelu<<<32 * H, 256, 0, stream>>>(cw, cb, Y, H);

  k_actquant_nchw<<<32 * 128, 256, 0, stream>>>(Y, XQ2, IS2);

  k_gemm_iu8<<<dim3(M / 32, D / 256), 256, 0, stream>>>(XQ2, WQ2, IS2, WS2, b2, out,
                                                        H, D, 0);
}
